// TFWindowAttention3D_81982335746628
// MI455X (gfx1250) — matrix operand order repaired and checked by reading, not compiled, not run on hardware
//
#include <hip/hip_runtime.h>
#include <hip/hip_bf16.h>

// ---------------- types ----------------
typedef __bf16 bf16;
typedef bf16  v16bf __attribute__((ext_vector_type(16)));
typedef float v8f   __attribute__((ext_vector_type(8)));

#define NTOK 98
#define NPAD 112
#define DIMC 128
#define NH   4
#define HD   32
#define NWIN 64
#define LSTR 136    // padded LDS row stride (bf16 elems): 272 B = 68 banks -> conflict-free
#define SSTR 113    // padded score row stride (f32 elems): odd -> conflict-free
#define SCALE_F 0.17677669529663687f   // 32^-0.5

__device__ inline bf16 f2bf(float f) { return (bf16)f; }

// A fragment: M x K = 16 x 32 bf16, source row-major (K contiguous), stride in elements.
// lanes 0-15: row = lane, K = 0..15; lanes 16-31: row = lane-16, K = 16..31.
__device__ inline v16bf load_a_frag(const bf16* base, int stride, int row0, int k0, int lane) {
    int r  = lane & 15;
    int hi = lane >> 4;
    const bf16* p = base + (row0 + r) * stride + k0 + hi * 16;
    v16bf f;
    uint4* d = reinterpret_cast<uint4*>(&f);
    d[0] = *reinterpret_cast<const uint4*>(p);
    d[1] = *reinterpret_cast<const uint4*>(p + 8);
    return f;
}

// B fragment: K x N = 32 x 16 bf16. We store B^T row-major (row = N index, K contiguous).
// lanes 0-15: col = lane, K = 0..15; lanes 16-31: col = lane-16, K = 16..31.
__device__ inline v16bf load_b_frag(const bf16* base, int stride, int n0, int k0, int lane) {
    int n  = lane & 15;
    int hi = lane >> 4;
    const bf16* p = base + (n0 + n) * stride + k0 + hi * 16;
    v16bf f;
    uint4* d = reinterpret_cast<uint4*>(&f);
    d[0] = *reinterpret_cast<const uint4*>(p);
    d[1] = *reinterpret_cast<const uint4*>(p + 8);
    return f;
}

__device__ inline v8f wmma_bf16f32(v16bf a, v16bf b, v8f c) {
    return __builtin_amdgcn_wmma_f32_16x16x32_bf16(false, a, false, b, (short)0, c, false, false);
}

// ---------------- prep kernels ----------------
__global__ __launch_bounds__(256) void prep_weights_kernel(
    const float* __restrict__ qkv_w,   // [128][384] row-major (in, out)
    const float* __restrict__ proj_w,  // [128][128]
    bf16* __restrict__ wqkvT,          // [384][128]  (out, in)
    bf16* __restrict__ projT)          // [128][128]  (out, in)
{
    int i = blockIdx.x * blockDim.x + threadIdx.x;
    if (i < 128 * 384) {
        int in = i / 384, o = i % 384;
        wqkvT[o * 128 + in] = f2bf(qkv_w[i]);
        return;
    }
    int j = i - 128 * 384;
    if (j < 128 * 128) {
        int in = j / 128, o = j % 128;
        projT[o * 128 + in] = f2bf(proj_w[j]);
    }
}

// bias[h][n][m] = table[REL_IDX(n,m)][h];  REL_IDX = (dz+1)*169 + (dy+6)*13 + (dx+6)
__global__ __launch_bounds__(256) void prep_bias_kernel(
    const float* __restrict__ table,   // [507][4]
    float* __restrict__ bias)          // [4][98][98]
{
    int i = blockIdx.x * blockDim.x + threadIdx.x;
    if (i >= NH * NTOK * NTOK) return;
    int h  = i / (NTOK * NTOK);
    int nm = i % (NTOK * NTOK);
    int n = nm / NTOK, m = nm % NTOK;
    int az = n / 49, ay = (n / 7) % 7, ax = n % 7;
    int bz = m / 49, by = (m / 7) % 7, bx = m % 7;
    int idx = (az - bz + 1) * 169 + (ay - by + 6) * 13 + (ax - bx + 6);
    bias[i] = table[idx * NH + h];
}

// ---------------- fused window-attention kernel ----------------
__global__ __launch_bounds__(256) void window_attn_kernel(
    const float* __restrict__ x,       // [4096][98][128]
    const float* __restrict__ mask,    // [64][98][98]
    const bf16*  __restrict__ wqkvT,   // [384][128] bf16
    const float* __restrict__ qkv_b,   // [384]
    const bf16*  __restrict__ projT,   // [128][128] bf16
    const float* __restrict__ proj_b,  // [128]
    const float* __restrict__ bias,    // [4][98][98]
    float* __restrict__ out)           // [4096][98][128]
{
    extern __shared__ char smem[];
    bf16*  Xs = (bf16*)smem;                        // [112][LSTR]  x, later P (softmax probs)
    bf16*  Qs = Xs + NPAD * LSTR;                   // [112][LSTR]  scaled Q (all heads)
    bf16*  Ks = Qs + NPAD * LSTR;                   // [112][LSTR]  K (all heads)
    bf16*  Vt = Ks + NPAD * LSTR;                   // [128][LSTR]  V^T [d][m], m padded w/ 0
    float* Ss = (float*)(Vt + DIMC * LSTR);         // [112][SSTR]  attention logits (one head)
    bf16*  Os = (bf16*)(Ss + NPAD * SSTR);          // [112][LSTR]  attention output (all heads)

    const int b    = blockIdx.x;
    const int tid  = threadIdx.x;
    const int lane = tid & 31;
    const int wave = tid >> 5;
    const int c16  = lane & 15;
    const int hi   = lane >> 4;

    // ---- zero V^T pad columns (tokens 98..127) so AV WMMA never sees garbage B ----
    for (int i = tid; i < DIMC * (DIMC - NTOK); i += 256) {
        int d = i / (DIMC - NTOK);
        int m = NTOK + i % (DIMC - NTOK);
        Vt[d * LSTR + m] = f2bf(0.0f);
    }
    // ---- stage x -> LDS as bf16 (float4 loads, 8B packed stores) ----
    const float* xb = x + (size_t)b * NTOK * DIMC;
    for (int i = tid; i < NTOK * DIMC / 4; i += 256) {
        float4 v = reinterpret_cast<const float4*>(xb)[i];
        int row = (i * 4) / DIMC, col = (i * 4) % DIMC;
        bf16 t[4] = {f2bf(v.x), f2bf(v.y), f2bf(v.z), f2bf(v.w)};
        *reinterpret_cast<uint2*>(Xs + row * LSTR + col) = *reinterpret_cast<uint2*>(t);
    }
    __syncthreads();

    // ---- Stage A: QKV = X @ Wqkv  (C tiles 7x24, K = 128) ----
    for (int job0 = wave; job0 < 7 * 24; job0 += 8) {
        const int job = __builtin_amdgcn_readfirstlane(job0);   // wave-uniform -> SALU
        const int mt = job / 24, nt = job % 24;
        v8f acc = {0.f, 0.f, 0.f, 0.f, 0.f, 0.f, 0.f, 0.f};
#pragma unroll
        for (int kk = 0; kk < 4; ++kk) {
            v16bf a = load_a_frag(Xs,    LSTR, mt * 16, kk * 32, lane);
            v16bf w = load_b_frag(wqkvT, DIMC, nt * 16, kk * 32, lane);
            acc = wmma_bf16f32(a, w, acc);
        }
        int colg = nt * 16 + c16;            // global qkv column 0..383
        float bv = qkv_b[colg];
        int row0 = mt * 16 + hi * 8;

        if (nt < 8) {                         // ---- Q region (scaled) ---- (scalar branch)
            bf16* q = Qs + colg;
            if (mt < 6) {
#pragma unroll
                for (int r = 0; r < 8; ++r)
                    q[(row0 + r) * LSTR] = f2bf((acc[r] + bv) * SCALE_F);
            } else if (hi == 0) {             // rows 96,97 only
                q[96 * LSTR] = f2bf((acc[0] + bv) * SCALE_F);
                q[97 * LSTR] = f2bf((acc[1] + bv) * SCALE_F);
            }
        } else if (nt < 16) {                 // ---- K region ----
            bf16* k = Ks + (colg - 128);
            if (mt < 6) {
#pragma unroll
                for (int r = 0; r < 8; ++r)
                    k[(row0 + r) * LSTR] = f2bf(acc[r] + bv);
            } else if (hi == 0) {
                k[96 * LSTR] = f2bf(acc[0] + bv);
                k[97 * LSTR] = f2bf(acc[1] + bv);
            }
        } else {                              // ---- V region, stored transposed [d][m] ----
            bf16* v = Vt + (colg - 256) * LSTR;
            if (mt < 6) {
#pragma unroll
                for (int r = 0; r < 8; ++r)
                    v[row0 + r] = f2bf(acc[r] + bv);
            } else if (hi == 0) {
                v[96] = f2bf(acc[0] + bv);
                v[97] = f2bf(acc[1] + bv);
            }
        }
    }
    __syncthreads();

    const float* maskb = mask + (size_t)(b % NWIN) * NTOK * NTOK;

    for (int h = 0; h < NH; ++h) {
        const float* biash = bias + (size_t)h * NTOK * NTOK;
        // ---- S = Q_h @ K_h^T + bias + mask   (49 tiles, K = 32 -> one WMMA) ----
        for (int job0 = wave; job0 < 49; job0 += 8) {
            const int job = __builtin_amdgcn_readfirstlane(job0);
            const int it = job / 7, jt = job % 7;
            v16bf a = load_a_frag(Qs, LSTR, it * 16, h * HD, lane);
            v16bf k = load_b_frag(Ks, LSTR, jt * 16, h * HD, lane);
            v8f acc = {0.f, 0.f, 0.f, 0.f, 0.f, 0.f, 0.f, 0.f};
            acc = wmma_bf16f32(a, k, acc);
            int m   = jt * 16 + c16;
            int mc  = m < NTOK - 1 ? m : NTOK - 1;   // clamped gather col
            bool mok = (m < NTOK);
            int n0  = it * 16 + hi * 8;
            float* srow = Ss + n0 * SSTR + m;
#pragma unroll
            for (int r = 0; r < 8; ++r) {
                int n  = n0 + r;
                int nc = n < NTOK - 1 ? n : NTOK - 1;
                float v = acc[r] + biash[nc * NTOK + mc] + maskb[nc * NTOK + mc];
                srow[r * SSTR] = (mok && n < NTOK) ? v : -1e30f;   // v_cndmask, no branch
            }
        }
        __syncthreads();

        // ---- row softmax -> P (bf16) overlaid on Xs, cols padded to 128 with 0 ----
        if (tid < NPAD) {
            const float* srow = Ss + tid * SSTR;
            float mx = -1e30f;
            for (int m2 = 0; m2 < NTOK; ++m2) mx = fmaxf(mx, srow[m2]);
            float sum = 0.f;
            for (int m2 = 0; m2 < NTOK; ++m2) sum += __expf(srow[m2] - mx);
            float inv = 1.0f / sum;
            bf16* prow = Xs + tid * LSTR;
            for (int m2 = 0; m2 < NTOK; ++m2) prow[m2] = f2bf(__expf(srow[m2] - mx) * inv);
            for (int m2 = NTOK; m2 < DIMC; ++m2) prow[m2] = f2bf(0.0f);
        }
        __syncthreads();

        // ---- O_h = P @ V_h   (7 x 2 tiles, K = 128 padded tokens) ----
        for (int job0 = wave; job0 < 14; job0 += 8) {
            const int job = __builtin_amdgcn_readfirstlane(job0);
            const int mt = job / 2, nt = job % 2;
            v8f acc = {0.f, 0.f, 0.f, 0.f, 0.f, 0.f, 0.f, 0.f};
#pragma unroll
            for (int kk = 0; kk < 4; ++kk) {
                v16bf p = load_a_frag(Xs, LSTR, mt * 16, kk * 32, lane);
                v16bf v = load_b_frag(Vt, LSTR, h * HD + nt * 16, kk * 32, lane);
                acc = wmma_bf16f32(p, v, acc);
            }
            bf16* o = Os + (mt * 16 + hi * 8) * LSTR + h * HD + nt * 16 + c16;
#pragma unroll
            for (int r = 0; r < 8; ++r)
                o[r * LSTR] = f2bf(acc[r]);
        }
        __syncthreads();   // Os/Xs stable before next head's softmax rewrites Xs
    }

    // ---- Y = O @ Wproj + b  -> global f32 (7 x 8 tiles, K = 128) ----
    float* outb = out + (size_t)b * NTOK * DIMC;
    for (int job0 = wave; job0 < 7 * 8; job0 += 8) {
        const int job = __builtin_amdgcn_readfirstlane(job0);
        const int mt = job / 8, nt = job % 8;
        v8f acc = {0.f, 0.f, 0.f, 0.f, 0.f, 0.f, 0.f, 0.f};
#pragma unroll
        for (int kk = 0; kk < 4; ++kk) {
            v16bf o = load_a_frag(Os,    LSTR, mt * 16, kk * 32, lane);
            v16bf w = load_b_frag(projT, DIMC, nt * 16, kk * 32, lane);
            acc = wmma_bf16f32(o, w, acc);
        }
        int col = nt * 16 + c16;
        float bv = proj_b[col];
        float* y = outb + (mt * 16 + hi * 8) * DIMC + col;
        if (mt < 6) {
#pragma unroll
            for (int r = 0; r < 8; ++r)
                y[r * DIMC] = acc[r] + bv;
        } else if (hi == 0) {                 // rows 96,97 only
            y[0]    = acc[0] + bv;
            y[DIMC] = acc[1] + bv;
        }
    }
}

// ---------------- launch ----------------
extern "C" void kernel_launch(void* const* d_in, const int* in_sizes, int n_in,
                              void* d_out, int out_size, void* d_ws, size_t ws_size,
                              hipStream_t stream) {
    const float* x      = (const float*)d_in[0];
    const float* mask   = (const float*)d_in[1];
    const float* qkv_w  = (const float*)d_in[2];
    const float* qkv_b  = (const float*)d_in[3];
    const float* proj_w = (const float*)d_in[4];
    const float* proj_b = (const float*)d_in[5];
    const float* table  = (const float*)d_in[6];
    float* out = (float*)d_out;

    char* ws = (char*)d_ws;
    bf16*  wqkvT = (bf16*)ws;                         // 384*128*2 = 98304 B
    bf16*  projT = (bf16*)(ws + 98304);               // 32768 B
    float* bias  = (float*)(ws + 98304 + 32768);      // 153664 B

    prep_weights_kernel<<<(128 * 384 + 128 * 128 + 255) / 256, 256, 0, stream>>>(
        qkv_w, proj_w, wqkvT, projT);
    prep_bias_kernel<<<(NH * NTOK * NTOK + 255) / 256, 256, 0, stream>>>(table, bias);

    size_t smem = (size_t)NPAD * LSTR * 2 * 3   // Xs/P, Qs, Ks
                + (size_t)DIMC * LSTR * 2       // Vt
                + (size_t)NPAD * SSTR * 4       // Ss
                + (size_t)NPAD * LSTR * 2;      // Os   => 207,296 B (< 320 KB/WGP)
    (void)hipFuncSetAttribute((const void*)window_attn_kernel,
                              hipFuncAttributeMaxDynamicSharedMemorySize, (int)smem);

    window_attn_kernel<<<4096, 256, smem, stream>>>(
        x, mask, wqkvT, qkv_b, projT, proj_b, bias, out);
}
